// RPN_19035295055941
// MI455X (gfx1250) — matrix operand order repaired and checked by reading, not compiled, not run on hardware
//
#include <hip/hip_runtime.h>
#include <math.h>

typedef __attribute__((ext_vector_type(16))) _Float16 v16h;
typedef __attribute__((ext_vector_type(8)))  _Float16 v8h;
typedef __attribute__((ext_vector_type(8)))  float    v8f;

#define NEG_SCORE (-1000000000.0f)

// ---------- WMMA fragment loaders (LDS tiles are row-major [rows][32] f16) ----------
// A 16x32 f16: lanes 0-15 (row=lane): elems 0..7 = K0..7, 8..15 = K16..23
//              lanes 16-31 (row=lane-16): elems 0..7 = K8..15, 8..15 = K24..31
__device__ __forceinline__ v16h frag_a(const _Float16* base, int row, int lane) {
  const _Float16* p = base + (row + (lane & 15)) * 32 + ((lane >> 4) << 3);
  v8h lo = *(const v8h*)(p);
  v8h hi = *(const v8h*)(p + 16);
  v16h r;
#pragma unroll
  for (int i = 0; i < 8; ++i) { r[i] = lo[i]; r[i + 8] = hi[i]; }
  return r;
}
// B 32x16 f16 stored as BT[n][k]: lane holds column n=lane&15, same K order as A:
// elems 0..7 = K(lane>>4)*8.., 8..15 = K16+(lane>>4)*8..
__device__ __forceinline__ v16h frag_b(const _Float16* base, int row, int lane) {
  const _Float16* p = base + (row + (lane & 15)) * 32 + ((lane >> 4) << 3);
  v8h lo = *(const v8h*)(p);
  v8h hi = *(const v8h*)(p + 16);
  v16h r;
#pragma unroll
  for (int i = 0; i < 8; ++i) { r[i] = lo[i]; r[i + 8] = hi[i]; }
  return r;
}

// ---------- weight repack: OIHW f32 -> [co][tap][ci] f16 ----------
__global__ void k_wt3x3(const float* __restrict__ w, _Float16* __restrict__ wt) {
  int t = blockIdx.x * blockDim.x + threadIdx.x;
  if (t >= 512 * 4608) return;
  int co = t / 4608;
  int r  = t - co * 4608;
  int tap = r >> 9;        // r / 512
  int ci  = r & 511;
  wt[t] = (_Float16)w[co * 4608 + ci * 9 + tap];
}

// ---------- pack 1x1 weights: rows 0..17 = w_cls, 18..53 = w_reg, 54..63 = 0 ----------
__global__ void k_wc(const float* __restrict__ wcls, const float* __restrict__ wreg,
                     _Float16* __restrict__ wc) {
  int t = blockIdx.x * blockDim.x + threadIdx.x;
  if (t >= 64 * 512) return;
  int n = t >> 9, ci = t & 511;
  float v = 0.0f;
  if (n < 18) v = wcls[n * 512 + ci];
  else if (n < 54) v = wreg[(n - 18) * 512 + ci];
  wc[t] = (_Float16)v;
}

// ---------- 3x3 conv 512->512 + bias + ReLU, WMMA f16, output NHWC f16 ----------
// grid: (256 [= b*32 + row-pair], 8 [co tiles]); block 256 = 8 waves.
// Block M-tile = 128 (two image rows). A LDS = two 66-row stripes (xi = x+1,
// xx in [-1,64], edge rows constant zero). Per-wave tile = 32x32 (4 accums):
// each fragment quad (a0,a1,b0,b1) feeds 4 WMMAs -> 2 ds_load_b128 per WMMA.
// Three dx taps reuse each A stripe -> 12 WMMAs per barrier pair.
__global__ __launch_bounds__(256) void k_conv3(const float* __restrict__ in,
    const _Float16* __restrict__ wt, const float* __restrict__ bias,
    _Float16* __restrict__ feat) {
  __shared__ __align__(16) _Float16 As[2 * 66 * 32];    // [stripe][xi][ci]
  __shared__ __align__(16) _Float16 Bs[3 * 64 * 32];    // [dx][n][ci]
  int tid = threadIdx.x;
  int lane = tid & 31, wv = tid >> 5;
  int m_off = (wv & 3) << 5;     // 0,32,64,96  (within M=128)
  int n_off = (wv >> 2) << 5;    // 0,32
  int b  = blockIdx.x >> 5;
  int y0 = (blockIdx.x & 31) << 1;     // rows y0, y0+1
  int co_base = blockIdx.y << 6;

  int xcol = tid & 63;   // staging: pixel x (A rows), output channel n (B rows)
  int seg  = tid >> 6;   // 0..3 -> 8-element K segment

  // zero-fill the zero-pad A rows (xi=0 -> xx=-1, xi=65 -> xx=64), both stripes
  if (tid < 16) {
    int s = tid >> 3, e = (tid >> 2) & 1, q = tid & 3;
    v8h z;
#pragma unroll
    for (int i = 0; i < 8; ++i) z[i] = (_Float16)0.0f;
    *(v8h*)(&As[s * 2112 + (e ? 65 : 0) * 32 + (q << 3)]) = z;
  }

  v8f acc[2][2] = {};
  const float* inb = in + (size_t)b * 512 * 4096;

#pragma unroll 1
  for (int dyi = 0; dyi < 3; ++dyi) {
    int yyb = y0 + dyi - 1;              // stripe 0 input row; stripe 1 = yyb+1
#pragma unroll 1
    for (int c16 = 0; c16 < 16; ++c16) {
      int cb = c16 << 5;
      // ---- stage both A stripes, rows xi=1..64 (xx = 0..63 always in range) ----
#pragma unroll
      for (int s = 0; s < 2; ++s) {
        int yy = yyb + s;
        bool yok = (yy >= 0) && (yy < 64);    // block-uniform -> scalar branch
        v8h av;
        if (yok) {
          const float* src = inb + (size_t)(cb + (seg << 3)) * 4096 + yy * 64 + xcol;
#pragma unroll
          for (int i = 0; i < 8; ++i) av[i] = (_Float16)src[(size_t)i * 4096];
        } else {
#pragma unroll
          for (int i = 0; i < 8; ++i) av[i] = (_Float16)0.0f;
        }
        *(v8h*)(&As[s * 2112 + (xcol + 1) * 32 + (seg << 3)]) = av;
      }
      // ---- stage B tiles for the 3 dx taps (contiguous f16 b128 copies) ----
#pragma unroll
      for (int dx = 0; dx < 3; ++dx) {
        const _Float16* srcb = wt + (size_t)(co_base + xcol) * 4608 +
                               (size_t)(dyi * 3 + dx) * 512 + cb + (seg << 3);
        *(v8h*)(&Bs[dx * 2048 + xcol * 32 + (seg << 3)]) = *(const v8h*)srcb;
      }
      __syncthreads();
      // ---- 12 WMMAs: 3 dx taps x (2 M-subtiles x 2 N-subtiles) ----
      const _Float16* Ab = As + ((m_off >> 6) * 2112);
      int xr = m_off & 63;
#pragma unroll
      for (int dx = 0; dx < 3; ++dx) {
        v16h a0 = frag_a(Ab, xr + dx, lane);
        v16h a1 = frag_a(Ab, xr + 16 + dx, lane);
        v16h b0 = frag_b(Bs + dx * 2048, n_off, lane);
        v16h b1 = frag_b(Bs + dx * 2048, n_off + 16, lane);
        acc[0][0] = __builtin_amdgcn_wmma_f32_16x16x32_f16(false, a0, false, b0, (short)0, acc[0][0], false, false);
        acc[0][1] = __builtin_amdgcn_wmma_f32_16x16x32_f16(false, a0, false, b1, (short)0, acc[0][1], false, false);
        acc[1][0] = __builtin_amdgcn_wmma_f32_16x16x32_f16(false, a1, false, b0, (short)0, acc[1][0], false, false);
        acc[1][1] = __builtin_amdgcn_wmma_f32_16x16x32_f16(false, a1, false, b1, (short)0, acc[1][1], false, false);
      }
      __syncthreads();
    }
  }
  // epilogue: C/D layout: lane&15 = N, lanes>=16 shift M by 8
  int nl = lane & 15, hiw = lane >> 4;
#pragma unroll
  for (int tm = 0; tm < 2; ++tm) {
#pragma unroll
    for (int tn = 0; tn < 2; ++tn) {
      v8f a = acc[tm][tn];
      int co = co_base + n_off + (tn << 4) + nl;
      float bv = bias[co];
#pragma unroll
      for (int r = 0; r < 8; ++r) {
        int mm = m_off + (tm << 4) + r + (hiw << 3);   // 0..127
        int y = y0 + (mm >> 6);
        int x = mm & 63;
        float v = a[r] + bv;
        v = v > 0.0f ? v : 0.0f;
        feat[((size_t)b * 4096 + y * 64 + x) * 512 + co] = (_Float16)v;
      }
    }
  }
}

// ---------- fused 1x1 convs (cls 18 + reg 36, padded to N=64), WMMA f16 ----------
// grid: 256 blocks (128 pixels each), block 256 = 8 waves, 32x32 per-wave tile
__global__ __launch_bounds__(256) void k_conv1(const _Float16* __restrict__ feat,
    const _Float16* __restrict__ wc, const float* __restrict__ bcls,
    const float* __restrict__ breg, float* __restrict__ cls, float* __restrict__ reg) {
  __shared__ __align__(16) _Float16 As[128 * 32];
  __shared__ __align__(16) _Float16 Bs[64 * 32];
  int tid = threadIdx.x;
  int lane = tid & 31, wv = tid >> 5;
  int m_off = (wv & 3) << 5;    // 0,32,64,96
  int n_off = (wv >> 2) << 5;   // 0,32
  int pix0 = blockIdx.x << 7;

  int arow = tid & 127, ahalf = tid >> 7;  // A staging: 16 halfs per thread
  int brow = tid & 63,  bseg  = tid >> 6;  // B staging: 8 halfs per thread

  v8f acc[2][2] = {};
#pragma unroll 1
  for (int c16 = 0; c16 < 16; ++c16) {
    int cb = c16 << 5;
    {
      const _Float16* src = feat + (size_t)(pix0 + arow) * 512 + cb + (ahalf << 4);
      *(v8h*)(&As[arow * 32 + (ahalf << 4)])     = *(const v8h*)(src);
      *(v8h*)(&As[arow * 32 + (ahalf << 4) + 8]) = *(const v8h*)(src + 8);
    }
    *(v8h*)(&Bs[brow * 32 + (bseg << 3)]) =
        *(const v8h*)(wc + (size_t)brow * 512 + cb + (bseg << 3));
    __syncthreads();
    v16h a0 = frag_a(As, m_off, lane);
    v16h a1 = frag_a(As, m_off + 16, lane);
    v16h b0 = frag_b(Bs, n_off, lane);
    v16h b1 = frag_b(Bs, n_off + 16, lane);
    acc[0][0] = __builtin_amdgcn_wmma_f32_16x16x32_f16(false, a0, false, b0, (short)0, acc[0][0], false, false);
    acc[0][1] = __builtin_amdgcn_wmma_f32_16x16x32_f16(false, a0, false, b1, (short)0, acc[0][1], false, false);
    acc[1][0] = __builtin_amdgcn_wmma_f32_16x16x32_f16(false, a1, false, b0, (short)0, acc[1][0], false, false);
    acc[1][1] = __builtin_amdgcn_wmma_f32_16x16x32_f16(false, a1, false, b1, (short)0, acc[1][1], false, false);
    __syncthreads();
  }
  int nl = lane & 15, hiw = lane >> 4;
#pragma unroll
  for (int tm = 0; tm < 2; ++tm) {
#pragma unroll
    for (int tn = 0; tn < 2; ++tn) {
      v8f a = acc[tm][tn];
      int n = n_off + (tn << 4) + nl;
#pragma unroll
      for (int r = 0; r < 8; ++r) {
        int pix = pix0 + m_off + (tm << 4) + r + (hiw << 3);
        int bb = pix >> 12, p = pix & 4095;
        if (n < 18)       cls[((size_t)bb * 4096 + p) * 18 + n]        = a[r] + bcls[n];
        else if (n < 54)  reg[((size_t)bb * 4096 + p) * 36 + (n - 18)] = a[r] + breg[n - 18];
      }
    }
  }
}

// ---------- softmax + anchor decode + clip + filter + sort-key ----------
__global__ void k_decode(const float* __restrict__ cls, const float* __restrict__ reg,
    const float* __restrict__ iminfo, float* __restrict__ boxes,
    unsigned long long* __restrict__ keys) {
  int t = blockIdx.x * blockDim.x + threadIdx.x;   // 8*65536 threads
  int b = t >> 16, i = t & 65535;
  if (b >= 8) return;
  if (i >= 36864) { keys[((size_t)b << 16) + i] = ~0ULL; return; }
  int p = i / 9, a = i - p * 9;
  int x = p & 63, y = p >> 6;
  float c0 = cls[((size_t)b * 4096 + p) * 18 + 2 * a];
  float c1 = cls[((size_t)b * 4096 + p) * 18 + 2 * a + 1];
  float mx = fmaxf(c0, c1);
  float e0 = expf(c0 - mx), e1 = expf(c1 - mx);
  float fg = e1 / (e0 + e1);
  int ai = a / 3, si = a - ai * 3;
  float asp = (ai == 0) ? 0.5f : (ai == 1 ? 1.0f : 2.0f);
  float sc  = (si == 0) ? 8.0f : (si == 1 ? 16.0f : 32.0f);
  float wsz = sqrtf(256.0f / asp) * sc;
  float hsz = sqrtf(256.0f * asp) * sc;
  float sx = x * 16.0f, sy = y * 16.0f;
  float ax1 = 7.5f - 0.5f * (wsz - 1.0f) + sx;
  float ay1 = 7.5f - 0.5f * (hsz - 1.0f) + sy;
  float ax2 = 7.5f + 0.5f * (wsz - 1.0f) + sx;
  float ay2 = 7.5f + 0.5f * (hsz - 1.0f) + sy;
  float wa = ax2 - ax1 + 1.0f, ha = ay2 - ay1 + 1.0f;
  float cxa = ax1 + 0.5f * wa, cya = ay1 + 0.5f * ha;
  const float* d = reg + ((size_t)b * 4096 + p) * 36 + a * 4;
  float cx = d[0] * wa + cxa;
  float cy = d[1] * ha + cya;
  float pw = expf(d[2]) * wa;
  float ph = expf(d[3]) * ha;
  float imh = iminfo[b * 3 + 0], imw = iminfo[b * 3 + 1], iscl = iminfo[b * 3 + 2];
  float x1 = cx - 0.5f * pw, y1 = cy - 0.5f * ph;
  float x2 = cx + 0.5f * pw, y2 = cy + 0.5f * ph;
  x1 = fminf(fmaxf(x1, 0.0f), imw - 1.0f);
  y1 = fminf(fmaxf(y1, 0.0f), imh - 1.0f);
  x2 = fminf(fmaxf(x2, 0.0f), imw - 1.0f);
  y2 = fminf(fmaxf(y2, 0.0f), imh - 1.0f);
  float msz = 16.0f * iscl;
  bool ok = (x2 - x1 + 1.0f >= msz) && (y2 - y1 + 1.0f >= msz);
  float score = ok ? fg : NEG_SCORE;
  float* bp = boxes + ((size_t)b * 36864 + i) * 4;
  bp[0] = x1; bp[1] = y1; bp[2] = x2; bp[3] = y2;
  // key: ascending sort == score descending, index ascending (lax.top_k semantics)
  unsigned sb  = __float_as_uint(score);
  unsigned ord = sb ^ ((sb >> 31) ? 0xFFFFFFFFu : 0x80000000u);
  unsigned dsc = ~ord;
  keys[((size_t)b << 16) + i] = ((unsigned long long)dsc << 32) | (unsigned)i;
}

// ---------- per-batch bitonic sort of 65536 u64 keys (L2-resident) ----------
__global__ __launch_bounds__(1024) void k_sort(unsigned long long* __restrict__ keys) {
  unsigned long long* k = keys + ((size_t)blockIdx.x << 16);
  const int N = 65536;
  for (int size = 2; size <= N; size <<= 1) {
    for (int stride = size >> 1; stride > 0; stride >>= 1) {
      for (int t = threadIdx.x; t < N; t += 1024) {
        int partner = t ^ stride;
        if (partner > t) {
          bool up = ((t & size) == 0);
          unsigned long long a = k[t], bb = k[partner];
          if ((a > bb) == up) { k[t] = bb; k[partner] = a; }
        }
      }
      __syncthreads();
    }
  }
}

// ---------- gather top-6000 boxes in sorted order ----------
__global__ void k_gather(const unsigned long long* __restrict__ keys,
    const float* __restrict__ boxes, float* __restrict__ sbox) {
  int t = blockIdx.x * blockDim.x + threadIdx.x;
  if (t >= 8 * 6000) return;
  int b = t / 6000, i = t - b * 6000;
  unsigned idx = (unsigned)(keys[((size_t)b << 16) + i] & 0xFFFFFFFFu);
  const float* src = boxes + ((size_t)b * 36864 + idx) * 4;
  float* dst = sbox + ((size_t)b * 6000 + i) * 4;
  dst[0] = src[0]; dst[1] = src[1]; dst[2] = src[2]; dst[3] = src[3];
}

// ---------- greedy NMS over sorted boxes, 300 picks, write (b,x1,y1,x2,y2) ----------
__global__ __launch_bounds__(256) void k_nms(const float* __restrict__ sbox,
                                             float* __restrict__ out) {
  __shared__ int valid[6000];
  __shared__ float sb4[4];
  __shared__ int head;
  int tid = threadIdx.x;
  const float* bx = sbox + (size_t)blockIdx.x * 6000 * 4;
  float* o = out + (size_t)blockIdx.x * 300 * 5;
  for (int i = tid; i < 6000; i += 256) valid[i] = 1;
  if (tid == 0) head = 0;
  __syncthreads();
  for (int it = 0; it < 300; ++it) {
    if (tid == 0) {
      int h = head;
      while (h < 6000 && !valid[h]) ++h;
      head = h;
      int sel = (h < 6000) ? h : 0;   // reference argmax-of-all-NEG picks index 0
      if (h < 6000) valid[h] = 0;
      float bx1 = bx[sel * 4 + 0], by1 = bx[sel * 4 + 1];
      float bx2 = bx[sel * 4 + 2], by2 = bx[sel * 4 + 3];
      sb4[0] = bx1; sb4[1] = by1; sb4[2] = bx2; sb4[3] = by2;
      o[it * 5 + 0] = (float)blockIdx.x;
      o[it * 5 + 1] = bx1; o[it * 5 + 2] = by1;
      o[it * 5 + 3] = bx2; o[it * 5 + 4] = by2;
    }
    __syncthreads();
    float jx1 = sb4[0], jy1 = sb4[1], jx2 = sb4[2], jy2 = sb4[3];
    float a0 = (jx2 - jx1 + 1.0f) * (jy2 - jy1 + 1.0f);
    for (int i = tid; i < 6000; i += 256) {
      if (!valid[i]) continue;
      float x1 = fmaxf(jx1, bx[i * 4 + 0]);
      float y1 = fmaxf(jy1, bx[i * 4 + 1]);
      float x2 = fminf(jx2, bx[i * 4 + 2]);
      float y2 = fminf(jy2, bx[i * 4 + 3]);
      float inter = fmaxf(0.0f, x2 - x1 + 1.0f) * fmaxf(0.0f, y2 - y1 + 1.0f);
      float a1 = (bx[i * 4 + 2] - bx[i * 4 + 0] + 1.0f) *
                 (bx[i * 4 + 3] - bx[i * 4 + 1] + 1.0f);
      float iou = inter / (a0 + a1 - inter);
      if (iou > 0.7f) valid[i] = 0;
    }
    __syncthreads();
  }
}

extern "C" void kernel_launch(void* const* d_in, const int* in_sizes, int n_in,
                              void* d_out, int out_size, void* d_ws, size_t ws_size,
                              hipStream_t stream) {
  const float* input  = (const float*)d_in[0];
  // d_in[1] = gt_box (unused in forward)
  const float* iminfo = (const float*)d_in[2];
  const float* wfeat  = (const float*)d_in[3];
  const float* bfeat  = (const float*)d_in[4];
  const float* wcls   = (const float*)d_in[5];
  const float* bcls   = (const float*)d_in[6];
  const float* wreg   = (const float*)d_in[7];
  const float* breg   = (const float*)d_in[8];

  char* ws = (char*)d_ws;
  _Float16*           WT   = (_Float16*)(ws + 0);          //  4,718,592 B  [co][tap][ci] f16
  _Float16*           WC   = (_Float16*)(ws + 4718592);    //     65,536 B  [64][512] f16
  _Float16*           FEAT = (_Float16*)(ws + 4784128);    // 33,554,432 B  NHWC f16
  float*              CLS  = (float*)   (ws + 38338560);   //  2,359,296 B
  float*              REG  = (float*)   (ws + 40697856);   //  4,718,592 B
  float*              BOX  = (float*)   (ws + 45416448);   //  4,718,592 B
  unsigned long long* KEY  = (unsigned long long*)(ws + 50135040); // 4,194,304 B
  float*              SBX  = (float*)   (ws + 54329344);   //    768,000 B
  float*              OUT  = (float*)d_out;

  k_wt3x3 <<< (512 * 4608 + 255) / 256, 256, 0, stream >>> (wfeat, WT);
  k_wc    <<< (64 * 512 + 255) / 256,   256, 0, stream >>> (wcls, wreg, WC);
  k_conv3 <<< dim3(256, 8), 256, 0, stream >>> (input, WT, bfeat, FEAT);
  k_conv1 <<< 256, 256, 0, stream >>> (FEAT, WC, bcls, breg, CLS, REG);
  k_decode<<< (8 * 65536) / 256, 256, 0, stream >>> (CLS, REG, iminfo, BOX, KEY);
  k_sort  <<< 8, 1024, 0, stream >>> (KEY);
  k_gather<<< (8 * 6000 + 255) / 256, 256, 0, stream >>> (KEY, BOX, SBX);
  k_nms   <<< 8, 256, 0, stream >>> (SBX, OUT);
}
